// AlphaZeroNetwork_42691974922273
// MI455X (gfx1250) — matrix operand order repaired and checked by reading, not compiled, not run on hardware
//
#include <hip/hip_runtime.h>
#include <hip/hip_bf16.h>

// ---------------- constants from the reference ----------------
#define BATCH   512
#define NNODES  90
#define CH      256
#define ROWS    (BATCH * NNODES)      // 46080
#define NBLK    6
#define NEDGE   5
#define ASP     2086
#define ASPPAD  2176                  // 2086 padded to 128
#define NQ      64
#define PSIZE   (BATCH * ASP)         // 1068032
#define KCAT    (NEDGE * CH)          // 1280

// ---------------- bf16 helpers ----------------
typedef __attribute__((ext_vector_type(16))) __bf16 v16bf_t;
typedef __attribute__((ext_vector_type(8)))  float  v8f_t;

__device__ __forceinline__ unsigned short f2bf(float f) {
    unsigned int u = __builtin_bit_cast(unsigned int, f);
    u += 0x7fffu + ((u >> 16) & 1u);   // round-to-nearest-even
    return (unsigned short)(u >> 16);
}

// ---- weight transpose + pad + f32->bf16: dst[n*rowStride + k] = W[k*N + n] ----
__global__ void k_w2bfT(const float* __restrict__ src, unsigned short* __restrict__ dst,
                        int K, int N, int rowStride, int nRows) {
    int i = blockIdx.x * blockDim.x + threadIdx.x;
    if (i >= nRows * K) return;
    int n = i / K, k = i - n * K;
    unsigned short v = 0;
    if (n < N) v = f2bf(src[(size_t)k * N + n]);
    dst[(size_t)n * rowStride + k] = v;
}

// ---------------- WMMA GEMM ----------------
// A[M,K] bf16 row-major; Bt[Npad][K] bf16 (n-major, pre-transposed, Npad mult of 128)
// Block tile 128x128, wave tile 32x64 (8 WMMAs), double-buffered LDS.
// M must be a multiple of 128. D = act(A@B + bias (+Cf))
enum { GF_RELU = 1, GF_ACCUM = 2, GF_WF32 = 4, GF_WBF = 8 };

template <int FLAGS, bool NGUARD>
__global__ __launch_bounds__(256)
void gemm_bf16_wmma(const unsigned short* __restrict__ A,
                    const unsigned short* __restrict__ Bt,
                    const float* __restrict__ bias,
                    float* __restrict__ Cf,
                    unsigned short* __restrict__ Cb,
                    int M, int Nn, int K)
{
    __shared__ __align__(16) unsigned short As[2][128][40];   // 20480 B
    __shared__ __align__(16) unsigned short Bs[2][128][40];   // 20480 B

    const int tid  = threadIdx.x;
    const int m0   = blockIdx.y * 128;
    const int n0   = blockIdx.x * 128;
    const int lane = tid & 31;
    const int wave = tid >> 5;            // 8 waves (wave32)
    const int wm   = (wave >> 1) * 32;    // 0/32/64/96
    const int wn   = (wave & 1) * 64;     // 0/64
    const int mr   = lane & 15;
    const int kh   = lane >> 4;           // lane half; holds k = 16*kh + s in this code

    // cooperative-load mapping: each thread 2x uint4 per tile, unconditional
    const int ldRow = tid >> 1;           // 0..127
    const int ldCol = (tid & 1) * 16;     // 0/16

    const unsigned short* aSrc = A  + (size_t)(m0 + ldRow) * K + ldCol;
    const unsigned short* bSrc = Bt + (size_t)(n0 + ldRow) * K + ldCol;

    v8f_t acc[2][4] = {};
    const int nk = K >> 5;

    // preload tile 0
    {
        uint4 a0 = *(const uint4*)aSrc;
        uint4 a1 = *(const uint4*)(aSrc + 8);
        uint4 b0 = *(const uint4*)bSrc;
        uint4 b1 = *(const uint4*)(bSrc + 8);
        *(uint4*)&As[0][ldRow][ldCol]     = a0;
        *(uint4*)&As[0][ldRow][ldCol + 8] = a1;
        *(uint4*)&Bs[0][ldRow][ldCol]     = b0;
        *(uint4*)&Bs[0][ldRow][ldCol + 8] = b1;
    }
    __syncthreads();

    for (int t = 0; t < nk; ++t) {
        const int cur = t & 1;
        uint4 na0 = {}, na1 = {}, nb0 = {}, nb1 = {};
        const bool more = (t + 1 < nk);
        if (more) {
            const unsigned short* ap = aSrc + (size_t)(t + 1) * 32;
            const unsigned short* bp = bSrc + (size_t)(t + 1) * 32;
            na0 = *(const uint4*)ap;
            na1 = *(const uint4*)(ap + 8);
            nb0 = *(const uint4*)bp;
            nb1 = *(const uint4*)(bp + 8);
            if (t + 2 < nk) {
                __builtin_prefetch(ap + 32, 0, 3);
                __builtin_prefetch(bp + 32, 0, 3);
            }
        }

        // fragments: element s of lane half kh holds k = 16*kh + s in A and B
        union { v16bf_t v; unsigned int u[8]; } afr[2], bfr[4];
        #pragma unroll
        for (int mt = 0; mt < 2; ++mt) {
            const unsigned short* arow = &As[cur][wm + mt * 16 + mr][0];
            #pragma unroll
            for (int j = 0; j < 8; ++j) {
                int kb = 16 * kh + 2 * j;
                afr[mt].u[j] = *(const unsigned int*)(arow + kb);
            }
        }
        #pragma unroll
        for (int nt = 0; nt < 4; ++nt) {
            const unsigned short* brow = &Bs[cur][wn + nt * 16 + mr][0];
            #pragma unroll
            for (int j = 0; j < 8; ++j) {
                int kb = 16 * kh + 2 * j;
                bfr[nt].u[j] = *(const unsigned int*)(brow + kb);
            }
        }

        #pragma unroll
        for (int mt = 0; mt < 2; ++mt)
            #pragma unroll
            for (int nt = 0; nt < 4; ++nt)
                acc[mt][nt] = __builtin_amdgcn_wmma_f32_16x16x32_bf16(
                    false, afr[mt].v, false, bfr[nt].v, (short)0, acc[mt][nt], false, false);

        if (more) {
            *(uint4*)&As[cur ^ 1][ldRow][ldCol]     = na0;
            *(uint4*)&As[cur ^ 1][ldRow][ldCol + 8] = na1;
            *(uint4*)&Bs[cur ^ 1][ldRow][ldCol]     = nb0;
            *(uint4*)&Bs[cur ^ 1][ldRow][ldCol + 8] = nb1;
        }
        __syncthreads();
    }

    // epilogue: VGPR r -> M = r (lanes 0-15) / r+8 (lanes 16-31), N = lane%16
    const int nloc  = lane & 15;
    const int mhalf = kh * 8;
    #pragma unroll
    for (int mt = 0; mt < 2; ++mt) {
        #pragma unroll
        for (int nt = 0; nt < 4; ++nt) {
            #pragma unroll
            for (int r = 0; r < 8; ++r) {
                int gm = m0 + wm + mt * 16 + mhalf + r;
                int gn = n0 + wn + nt * 16 + nloc;
                if (NGUARD && gn >= Nn) continue;
                float v = acc[mt][nt][r] + bias[gn];
                size_t oi = (size_t)gm * Nn + gn;
                if (FLAGS & GF_ACCUM) v += Cf[oi];
                if (FLAGS & GF_RELU)  v = fmaxf(v, 0.f);
                if (FLAGS & GF_WF32)  Cf[oi] = v;
                if (FLAGS & GF_WBF)   Cb[oi] = f2bf(v);
            }
        }
    }
}

template <int FLAGS, bool NGUARD>
static void gemm_launch(const unsigned short* A, const unsigned short* Bt, const float* bias,
                        float* Cf, unsigned short* Cb, int M, int N, int K, hipStream_t s) {
    dim3 grid((N + 127) / 128, M / 128);
    gemm_bf16_wmma<FLAGS, NGUARD><<<grid, 256, 0, s>>>(A, Bt, bias, Cf, Cb, M, N, K);
}

// ---------------- stem: h = relu(x_perm @ Wstem + b), K=15 ----------------
__global__ void k_stem(const float* __restrict__ x, const float* __restrict__ w,
                       const float* __restrict__ b, float* __restrict__ hf,
                       unsigned short* __restrict__ hb) {
    int i = blockIdx.x * blockDim.x + threadIdx.x;
    if (i >= ROWS * CH) return;
    int row = i >> 8, co = i & 255;
    int bb = row / NNODES, n = row % NNODES;
    float acc = b[co];
    const float* xp = x + (size_t)bb * 15 * NNODES + n;
    #pragma unroll
    for (int ci = 0; ci < 15; ++ci)
        acc = fmaf(xp[ci * NNODES], w[ci * CH + co], acc);
    acc = fmaxf(acc, 0.f);
    hf[i] = acc;
    hb[i] = f2bf(acc);
}

// ---------------- fused neighbor masked means: out[row][e*256+c], e = blockIdx.y ----------------
struct EdgeTabs {
    const int*   idx[NEDGE];
    const float* mask[NEDGE];
    int          deg[NEDGE];
};

__global__ void k_aggmean_cat(const float* __restrict__ h, EdgeTabs et,
                              unsigned short* __restrict__ out) {
    int e = blockIdx.y;
    int i = blockIdx.x * blockDim.x + threadIdx.x;
    if (i >= ROWS * CH) return;
    int row = i >> 8, cc = i & 255;
    int bb = row / NNODES, n = row % NNODES;
    const int*   idx  = et.idx[e];
    const float* mask = et.mask[e];
    int deg = et.deg[e];
    float s = 0.f, den = 0.f;
    for (int j = 0; j < deg; ++j) {
        int id = idx[n * deg + j];
        float mk = mask[n * deg + j];
        int iid = id < 0 ? 0 : id;
        s += mk * h[((size_t)bb * NNODES + iid) * CH + cc];
        den += mk;
    }
    den = fmaxf(den, 1.f);
    out[(size_t)row * KCAT + e * CH + cc] = f2bf(s / den);
}

// ---------------- sum of the 5 edge biases ----------------
__global__ void k_bias5(const float* b0, const float* b1, const float* b2,
                        const float* b3, const float* b4, float* out) {
    int i = blockIdx.x * blockDim.x + threadIdx.x;
    if (i < CH) out[i] = b0[i] + b1[i] + b2[i] + b3[i] + b4[i];
}

// ---------------- layernorm over concat([h, agg]) (512 ch), one wave32 per row ----------------
__global__ __launch_bounds__(256)
void k_ln(const float* __restrict__ h, const float* __restrict__ agg,
          const float* __restrict__ scale, const float* __restrict__ bias,
          unsigned short* __restrict__ outbf) {
    int wid = threadIdx.x >> 5, lane = threadIdx.x & 31;
    int row = blockIdx.x * 8 + wid;
    if (row >= ROWS) return;
    const float* hr = h   + (size_t)row * CH;
    const float* ar = agg + (size_t)row * CH;
    float vals[16];
    float s = 0.f, sq = 0.f;
    #pragma unroll
    for (int i = 0; i < 16; ++i) {
        int c = i * 32 + lane;
        float v = (c < CH) ? hr[c] : ar[c - CH];
        vals[i] = v; s += v; sq += v * v;
    }
    #pragma unroll
    for (int m = 16; m >= 1; m >>= 1) {
        s  += __shfl_xor(s,  m, 32);
        sq += __shfl_xor(sq, m, 32);
    }
    float mu  = s * (1.f / 512.f);
    float var = sq * (1.f / 512.f) - mu * mu;
    float inv = rsqrtf(var + 1e-6f);
    unsigned short* orow = outbf + (size_t)row * 512;
    #pragma unroll
    for (int i = 0; i < 16; ++i) {
        int c = i * 32 + lane;
        orow[c] = f2bf((vals[i] - mu) * inv * scale[c] + bias[c]);
    }
}

// ---------------- value head ----------------
__global__ void k_vmean(const float* __restrict__ h, float* __restrict__ vin) {
    int i = blockIdx.x * blockDim.x + threadIdx.x;   // 512*256
    if (i >= BATCH * CH) return;
    int b = i >> 8, c = i & 255;
    float s = 0.f;
    for (int n = 0; n < NNODES; ++n)
        s += h[((size_t)b * NNODES + n) * CH + c];
    vin[i] = s * (1.f / (float)NNODES);
}

__global__ void k_v1(const float* __restrict__ vin, const float* __restrict__ w,
                     const float* __restrict__ b, float* __restrict__ vh) {
    int i = blockIdx.x * blockDim.x + threadIdx.x;   // 512*128
    if (i >= BATCH * 128) return;
    int bb = i >> 7, j = i & 127;
    float acc = b[j];
    const float* vr = vin + (size_t)bb * CH;
    for (int k = 0; k < CH; ++k)
        acc = fmaf(vr[k], w[k * 128 + j], acc);
    vh[i] = fmaxf(acc, 0.f);
}

__global__ void k_v2(const float* __restrict__ vh, const float* __restrict__ w,
                     const float* __restrict__ b, float* __restrict__ out) {
    int i = blockIdx.x * blockDim.x + threadIdx.x;   // 512*64
    if (i >= BATCH * NQ) return;
    int bb = i >> 6, j = i & 63;
    float acc = b[j];
    const float* vr = vh + (size_t)bb * 128;
    for (int k = 0; k < 128; ++k)
        acc = fmaf(vr[k], w[k * NQ + j], acc);
    out[i] = tanhf(acc);
}

// ==================== host side ====================
extern "C" void kernel_launch(void* const* d_in, const int* in_sizes, int n_in,
                              void* d_out, int out_size, void* d_ws, size_t ws_size,
                              hipStream_t stream) {
    (void)n_in; (void)out_size; (void)ws_size;

    // ---- classify input ordering: insertion (x first) vs jax-sorted (x last) ----
    const int*   eidx[NEDGE];
    const float* emask[NEDGE];
    const float *ew[NBLK][NEDGE], *eb[NBLK][NEDGE];
    const float *lns[NBLK], *lnb[NBLK], *m1w[NBLK], *m1b[NBLK], *m2w[NBLK], *m2b[NBLK];
    const float *p1w, *p1b, *p2w, *p2b, *stw, *stb, *v1w, *v1b, *v2w, *v2b, *x;
    int degs[NEDGE];

    bool insertion = (in_sizes[0] == BATCH * 15 * NNODES);   // x == 691200 elems
    if (insertion) {
        x = (const float*)d_in[0];
        stw = (const float*)d_in[1]; stb = (const float*)d_in[2];
        for (int b = 0; b < NBLK; ++b) {
            int o = 3 + b * 16;
            for (int e = 0; e < NEDGE; ++e) {
                ew[b][e] = (const float*)d_in[o + 2 * e];
                eb[b][e] = (const float*)d_in[o + 2 * e + 1];
            }
            lns[b] = (const float*)d_in[o + 10]; lnb[b] = (const float*)d_in[o + 11];
            m1w[b] = (const float*)d_in[o + 12]; m1b[b] = (const float*)d_in[o + 13];
            m2w[b] = (const float*)d_in[o + 14]; m2b[b] = (const float*)d_in[o + 15];
        }
        p1w = (const float*)d_in[99];  p1b = (const float*)d_in[100];
        p2w = (const float*)d_in[101]; p2b = (const float*)d_in[102];
        v1w = (const float*)d_in[103]; v1b = (const float*)d_in[104];
        v2w = (const float*)d_in[105]; v2b = (const float*)d_in[106];
        for (int e = 0; e < NEDGE; ++e) {
            eidx[e]  = (const int*)d_in[107 + e];
            emask[e] = (const float*)d_in[112 + e];
            degs[e]  = in_sizes[107 + e] / NNODES;
        }
    } else {
        for (int e = 0; e < NEDGE; ++e) {
            eidx[e]  = (const int*)d_in[e];
            emask[e] = (const float*)d_in[5 + e];
            degs[e]  = in_sizes[e] / NNODES;
        }
        for (int b = 0; b < NBLK; ++b) {
            int o = 10 + b * 16;
            for (int e = 0; e < NEDGE; ++e) {
                eb[b][e] = (const float*)d_in[o + 2 * e];
                ew[b][e] = (const float*)d_in[o + 2 * e + 1];
            }
            lnb[b] = (const float*)d_in[o + 10]; lns[b] = (const float*)d_in[o + 11];
            m1b[b] = (const float*)d_in[o + 12]; m1w[b] = (const float*)d_in[o + 13];
            m2b[b] = (const float*)d_in[o + 14]; m2w[b] = (const float*)d_in[o + 15];
        }
        p1b = (const float*)d_in[106]; p1w = (const float*)d_in[107];
        p2b = (const float*)d_in[108]; p2w = (const float*)d_in[109];
        stb = (const float*)d_in[110]; stw = (const float*)d_in[111];
        v1b = (const float*)d_in[112]; v1w = (const float*)d_in[113];
        v2b = (const float*)d_in[114]; v2w = (const float*)d_in[115];
        x   = (const float*)d_in[116];
    }

    // ---- carve workspace ----
    char* ws = (char*)d_ws;
    size_t off = 0;
    auto carve = [&](size_t bytes) -> char* {
        char* p = ws + off;
        off = (off + bytes + 255) & ~(size_t)255;
        return p;
    };
    float*          hf32    = (float*)         carve((size_t)ROWS * CH * 4);
    unsigned short* hbf     = (unsigned short*)carve((size_t)ROWS * CH * 2);
    float*          aggf    = (float*)         carve((size_t)ROWS * CH * 4);
    unsigned short* meancat = (unsigned short*)carve((size_t)ROWS * KCAT * 2);
    unsigned short* xcatbf  = (unsigned short*)carve((size_t)ROWS * 512 * 2);
    unsigned short* t1bf    = (unsigned short*)carve((size_t)ROWS * CH * 2);
    unsigned short* pactbf  = (unsigned short*)carve((size_t)ROWS * 32 * 2);
    float*          vin     = (float*)         carve((size_t)BATCH * CH * 4);
    float*          vh      = (float*)         carve((size_t)BATCH * 128 * 4);
    float*          ebias   = (float*)         carve((size_t)CH * 4);
    unsigned short* wbf     = (unsigned short*)carve((size_t)9445376 * 2);

    // transposed/padded bf16 weight pointers
    unsigned short *ewtb[NBLK], *m1wb[NBLK], *m2wb[NBLK], *p1wb, *p2wb;
    size_t wo = 0;
    for (int b = 0; b < NBLK; ++b) {
        ewtb[b] = wbf + wo; wo += (size_t)CH * KCAT;       // stacked [256][1280]
        m1wb[b] = wbf + wo; wo += (size_t)CH * 512;        // [256][512]
        m2wb[b] = wbf + wo; wo += (size_t)CH * CH;         // [256][256]
    }
    p1wb = wbf + wo; wo += (size_t)128 * CH;               // [128][256] (32 padded to 128)
    p2wb = wbf + wo; wo += (size_t)ASPPAD * (NNODES * 32); // [2176][2880]

    auto wconv = [&](const float* src, unsigned short* dst, int K, int N,
                     int rowStride, int nRows) {
        long long tot = (long long)nRows * K;
        k_w2bfT<<<(unsigned)((tot + 255) / 256), 256, 0, stream>>>(src, dst, K, N,
                                                                   rowStride, nRows);
    };

    // ---- convert + transpose weights to bf16 (n-major, padded) ----
    for (int b = 0; b < NBLK; ++b) {
        for (int e = 0; e < NEDGE; ++e)
            wconv(ew[b][e], ewtb[b] + e * CH, CH, CH, KCAT, CH);  // column-block e of [256][1280]
        wconv(m1w[b], m1wb[b], 512, CH, 512, CH);
        wconv(m2w[b], m2wb[b], CH, CH, CH, CH);
    }
    wconv(p1w, p1wb, CH, 32, CH, 128);
    wconv(p2w, p2wb, NNODES * 32, ASP, NNODES * 32, ASPPAD);

    // ---- stem ----
    k_stem<<<(ROWS * CH + 255) / 256, 256, 0, stream>>>(x, stw, stb, hf32, hbf);

    // edge tables (uniform per blockIdx.y inside the kernel)
    EdgeTabs et;
    for (int e = 0; e < NEDGE; ++e) { et.idx[e] = eidx[e]; et.mask[e] = emask[e]; et.deg[e] = degs[e]; }
    dim3 aggGrid((ROWS * CH + 255) / 256, NEDGE);

    // ---- GNN blocks ----
    for (int b = 0; b < NBLK; ++b) {
        k_bias5<<<1, 256, 0, stream>>>(eb[b][0], eb[b][1], eb[b][2], eb[b][3], eb[b][4], ebias);
        k_aggmean_cat<<<aggGrid, 256, 0, stream>>>(hf32, et, meancat);
        // fused 5-edge message GEMM: [46080,1280] @ [1280,256]
        gemm_launch<GF_WF32, false>(meancat, ewtb[b], ebias, aggf, nullptr,
                                    ROWS, CH, KCAT, stream);
        k_ln<<<ROWS / 8, 256, 0, stream>>>(hf32, aggf, lns[b], lnb[b], xcatbf);
        gemm_launch<GF_RELU | GF_WBF, false>(xcatbf, m1wb[b], m1b[b], nullptr, t1bf,
                                             ROWS, CH, 512, stream);
        gemm_launch<GF_ACCUM | GF_WF32 | GF_WBF, false>(t1bf, m2wb[b], m2b[b], hf32, hbf,
                                                        ROWS, CH, CH, stream);
    }

    // ---- policy head ----
    gemm_launch<GF_RELU | GF_WBF, true>(hbf, p1wb, p1b, nullptr, pactbf,
                                        ROWS, 32, CH, stream);
    gemm_launch<GF_WF32, true>(pactbf, p2wb, p2b, (float*)d_out, nullptr,
                               BATCH, ASP, NNODES * 32, stream);

    // ---- value head ----
    k_vmean<<<(BATCH * CH + 255) / 256, 256, 0, stream>>>(hf32, vin);
    k_v1<<<(BATCH * 128 + 255) / 256, 256, 0, stream>>>(vin, v1w, v1b, vh);
    k_v2<<<(BATCH * NQ + 255) / 256, 256, 0, stream>>>(vh, v2w, v2b, (float*)d_out + PSIZE);
}
